// GCNBlock_63178968924655
// MI455X (gfx1250) — matrix operand order repaired and checked by reading, not compiled, not run on hardware
//
#include <hip/hip_runtime.h>
#include <hip/hip_bf16.h>

typedef __attribute__((ext_vector_type(16))) _Float16 v16h;
typedef __attribute__((ext_vector_type(4)))  _Float16 v4h;
typedef __attribute__((ext_vector_type(8)))  float    v8f;

#define F_IN  128
#define F_OUT 128

// ---------------------------------------------------------------------------
// Kernel 1: zero the output buffer (used as agg accumulator) and the counts.
// ---------------------------------------------------------------------------
__global__ void init_zero(float* __restrict__ out, long long n_out,
                          float* __restrict__ cnts, long long n_cnt) {
    long long i = (long long)blockIdx.x * blockDim.x + threadIdx.x;
    long long stride = (long long)gridDim.x * blockDim.x;
    for (long long k = i; k < n_out; k += stride) out[k] = 0.0f;
    for (long long k = i; k < n_cnt; k += stride) cnts[k] = 0.0f;
}

// ---------------------------------------------------------------------------
// Kernel 2: per-edge weighted scatter. One wave32 per edge; lane l moves
// float4 l of the 128-float source row. Hardware f32 atomics into L2-resident
// d_out (pos half cols 0..127, neg half cols 128..255).
// ---------------------------------------------------------------------------
__global__ void scatter_edges(const float* __restrict__ x,
                              const long long* __restrict__ ei,   // [2,E] int64
                              const float* __restrict__ ea,       // [E]
                              float* __restrict__ out,            // [N,256] agg sums
                              float* __restrict__ cnt_pos,
                              float* __restrict__ cnt_neg,
                              int E) {
    int wid  = (int)(((long long)blockIdx.x * blockDim.x + threadIdx.x) >> 5);
    int lane = threadIdx.x & 31;
    if (wid >= E) return;
    float a = ea[wid];
    if (a == 0.0f) return;                 // contributes to neither sign
    long long s = ei[wid];
    long long d = ei[(long long)E + wid];
    float w = fabsf(a);

    const float4* xr = (const float4*)(x + s * (long long)F_IN);
    float4 v = xr[lane];                   // coalesced 512B per edge

    float* base = out + d * 256LL + (a > 0.0f ? 0 : F_OUT) + lane * 4;
    unsafeAtomicAdd(base + 0, v.x * w);
    unsafeAtomicAdd(base + 1, v.y * w);
    unsafeAtomicAdd(base + 2, v.z * w);
    unsafeAtomicAdd(base + 3, v.w * w);
    if (lane == 0)
        unsafeAtomicAdd((a > 0.0f ? cnt_pos : cnt_neg) + d, 1.0f);
}

// ---------------------------------------------------------------------------
// Kernel 3: fused normalize + dual GEMM + bias + ReLU via WMMA f16->f32.
//   out[:, s*128 : s*128+128] = relu( (agg_s / max(cnt_s,1)) @ W_l^T
//                                      + x @ W_r^T + b )
// blockIdx.y = side (0 pos, 1 neg). Block = 256 threads = 8 waves; each wave
// owns a 16-row node tile and the full 128 output columns of its side, so it
// reads its agg rows entirely before overwriting them (no cross-wave hazard).
// Weights are staged f32->f16 into LDS once per block.
// ---------------------------------------------------------------------------
__global__ void __launch_bounds__(256) wmma_fused_gemm(
    const float* __restrict__ x,         // [N,128]
    float* __restrict__ out,             // [N,256]; holds agg sums on entry
    const float* __restrict__ cnt_pos,
    const float* __restrict__ cnt_neg,
    const float* __restrict__ w_pos_l, const float* __restrict__ w_pos_r,
    const float* __restrict__ b_pos,
    const float* __restrict__ w_neg_l, const float* __restrict__ w_neg_r,
    const float* __restrict__ b_neg,
    int N) {
    __shared__ _Float16 ldsL[F_OUT * F_IN];   // 32 KB
    __shared__ _Float16 ldsR[F_OUT * F_IN];   // 32 KB

    const int side = blockIdx.y;
    const float* wl   = side ? w_neg_l : w_pos_l;
    const float* wr   = side ? w_neg_r : w_pos_r;
    const float* bias = side ? b_neg   : b_pos;
    const float* cnt  = side ? cnt_neg : cnt_pos;

    // Cooperative f32 -> f16 weight staging into LDS.
    for (int i = threadIdx.x; i < (F_OUT * F_IN) / 4; i += 256) {
        float4 l4 = ((const float4*)wl)[i];
        float4 r4 = ((const float4*)wr)[i];
        v4h hl = { (_Float16)l4.x, (_Float16)l4.y, (_Float16)l4.z, (_Float16)l4.w };
        v4h hr = { (_Float16)r4.x, (_Float16)r4.y, (_Float16)r4.z, (_Float16)r4.w };
        *(v4h*)&ldsL[4 * i] = hl;
        *(v4h*)&ldsR[4 * i] = hr;
    }
    __syncthreads();

    const int wave = threadIdx.x >> 5;
    const int lane = threadIdx.x & 31;
    const int tile = blockIdx.x * 8 + wave;
    const int nTiles = (N + 15) >> 4;
    if (tile >= nTiles) return;            // wave-uniform: EXEC stays full

    const int half = lane >> 4;            // 0: lanes 0-15, 1: lanes 16-31
    const int lm   = lane & 15;
    int row = tile * 16 + lm;
    if (row >= N) row = N - 1;             // clamped lanes never store
    const float* xrow = x   + (long long)row * F_IN;
    const float* grow = out + (long long)row * 256 + side * F_OUT;  // agg sums
    const float  scale = 1.0f / fmaxf(cnt[row], 1.0f);

    v8f acc[8] = {};                       // 8 col-tiles x (16x16 f32)

    for (int ks = 0; ks < 4; ++ks) {       // K = 128 in 4 steps of 32
        // A fragments (16x32 f16): lane covers K [k0,k0+16)
        const int k0 = ks * 32 + half * 16;
        v16h a_x, a_g;
#pragma unroll
        for (int j = 0; j < 8; ++j) {
            a_x[j]     = (_Float16)xrow[k0 + j];
            a_x[j + 8] = (_Float16)xrow[k0 + 8 + j];
            a_g[j]     = (_Float16)(grow[k0 + j] * scale);
            a_g[j + 8] = (_Float16)(grow[k0 + 8 + j] * scale);
        }
        // B fragments (32x16 f16): lane = column c, contiguous 16 K per half.
        // B[k][c] = W[c][k], so column c of B is row c of W in LDS.
        const int kb = ks * 32 + half * 16;
#pragma unroll
        for (int t = 0; t < 8; ++t) {
            const int c0 = t * 16 + lm;
            v16h b_l = *(const v16h*)&ldsL[c0 * F_IN + kb];  // ds_load 32B
            v16h b_r = *(const v16h*)&ldsR[c0 * F_IN + kb];
            acc[t] = __builtin_amdgcn_wmma_f32_16x16x32_f16(
                false, a_x, false, b_r, (short)0, acc[t], false, false);
            acc[t] = __builtin_amdgcn_wmma_f32_16x16x32_f16(
                false, a_g, false, b_l, (short)0, acc[t], false, false);
        }
    }

    // Epilogue: bias + ReLU + store. C/D layout: VGPR i -> M = i + 8*half.
#pragma unroll
    for (int t = 0; t < 8; ++t) {
        const int col = t * 16 + lm;
        const float b = bias[col];
#pragma unroll
        for (int i = 0; i < 8; ++i) {
            const int r = tile * 16 + i + half * 8;
            if (r < N) {
                float v = acc[t][i] + b;
                out[(long long)r * 256 + side * F_OUT + col] = fmaxf(v, 0.0f);
            }
        }
    }
}

// ---------------------------------------------------------------------------
extern "C" void kernel_launch(void* const* d_in, const int* in_sizes, int n_in,
                              void* d_out, int out_size, void* d_ws, size_t ws_size,
                              hipStream_t stream) {
    const float*     x       = (const float*)d_in[0];
    const long long* ei      = (const long long*)d_in[1];   // int64 [2,E]
    const float*     ea      = (const float*)d_in[2];
    const float*     w_pos_l = (const float*)d_in[3];
    const float*     w_pos_r = (const float*)d_in[4];
    const float*     b_pos_r = (const float*)d_in[5];
    const float*     w_neg_l = (const float*)d_in[6];
    const float*     w_neg_r = (const float*)d_in[7];
    const float*     b_neg_r = (const float*)d_in[8];

    const int N = in_sizes[0] / F_IN;       // 50000
    const int E = in_sizes[2];              // 800000

    float* out     = (float*)d_out;         // [N,256]
    float* cnt_pos = (float*)d_ws;          // [N]
    float* cnt_neg = cnt_pos + N;           // [N]

    // 1) zero agg accumulators (in d_out) and counts (in d_ws)
    {
        long long n_out = (long long)N * 256;
        int blocks = (int)((n_out + 255) / 256);
        init_zero<<<blocks, 256, 0, stream>>>(out, n_out, cnt_pos, 2LL * N);
    }
    // 2) edge scatter: one wave per edge
    {
        long long threads = (long long)E * 32;
        int blocks = (int)((threads + 255) / 256);
        scatter_edges<<<blocks, 256, 0, stream>>>(x, ei, ea, out,
                                                  cnt_pos, cnt_neg, E);
    }
    // 3) fused normalize + GEMMs + bias + ReLU (WMMA)
    {
        int nTiles = (N + 15) / 16;
        dim3 grid((nTiles + 7) / 8, 2);
        wmma_fused_gemm<<<grid, 256, 0, stream>>>(x, out, cnt_pos, cnt_neg,
                                                  w_pos_l, w_pos_r, b_pos_r,
                                                  w_neg_l, w_neg_r, b_neg_r, N);
    }
}
